// LogisticMixture_70085276336360
// MI455X (gfx1250) — matrix operand order repaired and checked by reading, not compiled, not run on hardware
//
#include <hip/hip_runtime.h>
#include <math.h>

// ---------------------------------------------------------------------------
// LogisticMixture on MI455X (gfx1250, wave32, WMMA + async-to-LDS)
//
// Kernel 0: pack W_conv (f32) into bf16 B-fragment layout in d_ws (one pass).
// Kernel 1: per 128-pixel block:
//   - stage packed W image (57344B) global -> LDS via global_load_async_to_lds_b128
//     (ASYNCcnt, s_wait_asynccnt 0)
//   - GEMM: 8 waves x (16 pixels x 112 outputs), 56 v_wmma_f32_16x16x32_bf16/wave
//     A fragments packed with v_add_nc_u32 + v_perm_b32 (3 VALU per dword)
//   - phase 2: discretized-logistic-mixture log-prob, pairwise logsumexp merge
//     via wave32 shfl_xor, deterministic tree reduce -> per-block partial
// Kernel 2: fixed-order sum of 8 partials per batch image -> out[128]
// ---------------------------------------------------------------------------

typedef float  v8f   __attribute__((ext_vector_type(8)));
typedef __bf16 v16bf __attribute__((ext_vector_type(16)));

union BF16x16 {
    unsigned int u[8];
    v16bf        v;
};

#define D_IN     256   // feature depth (K)
#define N_PAD    112   // 100 outputs padded to 7*16
#define N_TILES  7
#define K_CHUNKS 8     // 256 / 32
#define PIX_PER_BLOCK 128
#define THREADS  256
#define W_DWORDS 14336 // 7 tiles * 8 chunks * 2 halves * 32 lanes * 4 dwords

// exact RNE pack (one-shot W pack kernel only)
__device__ __forceinline__ unsigned int pack_bf16_rne(float a, float b) {
    unsigned int ua = __float_as_uint(a);
    unsigned int ub = __float_as_uint(b);
    ua = (ua + 0x7FFFu + ((ua >> 16) & 1u)) >> 16;
    ub = (ub + 0x7FFFu + ((ub >> 16) & 1u)) >> 16;
    return ua | (ub << 16);
}

// fast pack for the GEMM hot loop: round-half-up + v_perm_b32
// result = { bf16(b), bf16(a) }  (a in low 16 bits)
__device__ __forceinline__ unsigned int pack_bf16_fast(float a, float b) {
    unsigned int ua = __float_as_uint(a) + 0x8000u;
    unsigned int ub = __float_as_uint(b) + 0x8000u;
    // sel 0x07060302: byte0=S1.b2, byte1=S1.b3 (a hi16), byte2=S0.b2, byte3=S0.b3 (b hi16)
    return __builtin_amdgcn_perm(ub, ua, 0x07060302u);
}

// log(sigmoid(z)) = min(z,0) - log(1+exp(-|z|)); 1+t is well-conditioned
__device__ __forceinline__ float logsigf(float z) {
    return fminf(z, 0.0f) - __logf(1.0f + __expf(-fabsf(z)));
}
__device__ __forceinline__ float softplusf(float z) {
    return fmaxf(z, 0.0f) + __logf(1.0f + __expf(-fabsf(z)));
}

// ---------------------------------------------------------------------------
// Kernel 0: pack W_conv into B-fragment bf16 layout + padded bias.
// dword index i = (((t*8+kk)*2 + h)*32 + lane)*4 + word
// fragment VGPR j = h*4+word holds K pair {kbase+16h+2word, +1},
// kbase = kk*32 + (lane>=16 ? 8 : 0), column n = t*16 + (lane&15).
// ---------------------------------------------------------------------------
__global__ __launch_bounds__(THREADS)
void lm_pack_kernel(const float* __restrict__ Wc,   // [256][100]
                    const float* __restrict__ bc,   // [100]
                    unsigned int* __restrict__ wpack,  // [14336]
                    float* __restrict__ bpad)          // [112]
{
    const int tid = threadIdx.x;
    const int i   = blockIdx.x * THREADS + tid;   // 56 blocks -> exactly 14336
    const int word   = i & 3;
    const int lane_s = (i >> 2) & 31;
    const int h      = (i >> 7) & 1;
    const int tk     = i >> 8;          // 0..55
    const int kk     = tk & 7;
    const int t      = tk >> 3;         // 0..6
    const int n      = t * 16 + (lane_s & 15);
    const int k      = kk * 32 + ((lane_s >> 4) << 3) + h * 16 + 2 * word;
    const float w0 = (n < 100) ? Wc[k * 100 + n]       : 0.0f;
    const float w1 = (n < 100) ? Wc[(k + 1) * 100 + n] : 0.0f;
    wpack[i] = pack_bf16_rne(w0, w1);
    if (blockIdx.x == 0 && tid < N_PAD)
        bpad[tid] = (tid < 100) ? bc[tid] : 0.0f;
}

// ---------------------------------------------------------------------------
// Kernel 1: fused GEMM + mixture log-likelihood per 128-pixel block.
// ---------------------------------------------------------------------------
__global__ __launch_bounds__(THREADS)
void lm_main_kernel(const float* __restrict__ x,        // [131072][256]
                    const float* __restrict__ value,    // [131072][3]
                    const unsigned int* __restrict__ wsrc, // [14336] packed W
                    const float* __restrict__ bsrc,     // [112] padded bias
                    float* __restrict__ partials)       // [1024]
{
    // LDS: phase1 = packed-W image (57344B); reused in phase2 as f32 params
    // [128 pixels][112].
    __shared__ unsigned int lds_buf[W_DWORDS];
    __shared__ float        lds_b[N_PAD];
    __shared__ float        red[PIX_PER_BLOCK];

    const int tid  = threadIdx.x;
    const int lane = tid & 31;
    const int w    = tid >> 5;          // wave id 0..7

    // ---- stage bias ----
    if (tid < N_PAD) lds_b[tid] = bsrc[tid];

    // ---- stage packed W image: global -> LDS, asynchronous (ASYNCcnt) ----
    // 14336 dwords = 14 sweeps of (256 threads x 16B) b128 transfers.
    {
        const unsigned int lds_base =
            (unsigned int)(unsigned long long)(void*)&lds_buf[0];
        const unsigned long long gbase =
            (unsigned long long)(const void*)wsrc;
        #pragma unroll
        for (int it = 0; it < 14; ++it) {
            const unsigned int       ldso = lds_base + (unsigned)(it * 4096 + tid * 16);
            const unsigned long long ga   = gbase + (unsigned long long)(it * 4096 + tid * 16);
            asm volatile("global_load_async_to_lds_b128 %0, %1, off"
                         :: "v"(ldso), "v"(ga) : "memory");
        }
        asm volatile("s_wait_asynccnt 0x0" ::: "memory");
    }
    __syncthreads();

    // =========================== Phase 1: GEMM ===========================
    // Wave w owns pixels [blockIdx.x*128 + w*16, +16).
    const int pix0 = blockIdx.x * PIX_PER_BLOCK + w * 16;
    const int row  = pix0 + (lane & 15);
    // A-fragment 16x32 bf16 layout: lanes>=16 start at K offset +8.
    const float* xrow = x + (size_t)row * D_IN + ((lane >> 4) << 3);

    v8f acc[N_TILES];
    #pragma unroll
    for (int t = 0; t < N_TILES; ++t)
        #pragma unroll
        for (int e = 0; e < 8; ++e) acc[t][e] = 0.0f;

    #pragma unroll
    for (int kk = 0; kk < K_CHUNKS; ++kk) {
        // Build A fragment for this 32-wide K chunk.
        const float* xp = xrow + kk * 32;
        float4 f0 = *(const float4*)(xp);        // K = kbase+0..3
        float4 f1 = *(const float4*)(xp + 4);    // K = kbase+4..7
        float4 f2 = *(const float4*)(xp + 16);   // K = kbase+16..19
        float4 f3 = *(const float4*)(xp + 20);   // K = kbase+20..23
        BF16x16 A;
        A.u[0] = pack_bf16_fast(f0.x, f0.y);
        A.u[1] = pack_bf16_fast(f0.z, f0.w);
        A.u[2] = pack_bf16_fast(f1.x, f1.y);
        A.u[3] = pack_bf16_fast(f1.z, f1.w);
        A.u[4] = pack_bf16_fast(f2.x, f2.y);
        A.u[5] = pack_bf16_fast(f2.z, f2.w);
        A.u[6] = pack_bf16_fast(f3.x, f3.y);
        A.u[7] = pack_bf16_fast(f3.z, f3.w);

        #pragma unroll
        for (int t = 0; t < N_TILES; ++t) {
            const uint4* bp0 =
                (const uint4*)&lds_buf[(((t * 8 + kk) * 2 + 0) * 32 + lane) * 4];
            const uint4* bp1 =
                (const uint4*)&lds_buf[(((t * 8 + kk) * 2 + 1) * 32 + lane) * 4];
            uint4 b0 = *bp0;
            uint4 b1 = *bp1;
            BF16x16 Bf;
            Bf.u[0] = b0.x; Bf.u[1] = b0.y; Bf.u[2] = b0.z; Bf.u[3] = b0.w;
            Bf.u[4] = b1.x; Bf.u[5] = b1.y; Bf.u[6] = b1.z; Bf.u[7] = b1.w;
            acc[t] = __builtin_amdgcn_wmma_f32_16x16x32_bf16(
                false, A.v, false, Bf.v, (short)0, acc[t], false, false);
        }
    }

    __syncthreads();   // everyone done reading W from LDS

    // Store params (+bias) as f32 into reused LDS: [pixel_local][112].
    // C layout: lanes 0-15: N=lane, M=vgpr; lanes 16-31: N=lane-16, M=8+vgpr.
    float* lds_f = reinterpret_cast<float*>(lds_buf);
    {
        const int prow0 = w * 16 + ((lane >> 4) << 3);
        const int ncol  = lane & 15;
        #pragma unroll
        for (int t = 0; t < N_TILES; ++t) {
            const int col  = t * 16 + ncol;
            const float bb = lds_b[col];
            #pragma unroll
            for (int v = 0; v < 8; ++v)
                lds_f[(prow0 + v) * N_PAD + col] = acc[t][v] + bb;
        }
    }
    __syncthreads();

    // ====================== Phase 2: mixture log-prob ======================
    // 2 threads per pixel; each handles 5 of 10 mixtures, merged via shfl_xor.
    const int   pL   = tid >> 1;
    const int   half = tid & 1;
    const int   gp   = blockIdx.x * PIX_PER_BLOCK + pL;
    const float y0 = value[gp * 3 + 0];
    const float y1 = value[gp * 3 + 1];
    const float y2 = value[gp * 3 + 2];
    const float tv0 = y0 * (2.0f / 255.0f) - 1.0f;
    const float tv1 = y1 * (2.0f / 255.0f) - 1.0f;

    float logitA[5], clpA[5];
    #pragma unroll
    for (int jj = 0; jj < 5; ++jj) {
        const int j = half * 5 + jj;
        const float* pp = &lds_f[pL * N_PAD + j * 10];
        const float logit = pp[0];
        float l0 = pp[1], l1 = pp[2], l2 = pp[3];
        const float s0 = pp[4], s1 = pp[5], s2 = pp[6];
        const float c0 = pp[7], c1 = pp[8], c2 = pp[9];
        l1 += tv0 * c0;
        l2 += tv0 * c1 + tv1 * c2;

        const float locs[3] = { l0, l1, l2 };
        const float sraw[3] = { s0, s1, s2 };
        const float ys[3]   = { y0, y1, y2 };
        float clp = 0.0f;
        #pragma unroll
        for (int c = 0; c < 3; ++c) {
            const float loc_t   = 127.5f * (locs[c] + 1.0f);
            const float scale_t = (softplusf(sraw[c]) + 9.11881966e-4f) * 127.5f;
            const float inv     = 1.0f / scale_t;
            const float pin  = (ys[c] + 0.5f - loc_t) * inv;
            const float minn = (ys[c] - 0.5f - loc_t) * inv;
            const float lcp = logsigf(pin);
            const float lcm = logsigf(minn);
            const float lsm = logsigf(-minn);
            const float diff = fminf(lcm - lcp, -1e-6f);
            const float lmid = lcp + log1pf(-__expf(diff));   // cancellation-sensitive
            const float lp = (ys[c] < 0.5f) ? lcp
                             : ((ys[c] > 254.5f) ? lsm : lmid);
            clp += lp;
        }
        logitA[jj] = logit;
        clpA[jj]   = clp;
    }

    // local logsumexp over 5 mixtures (logits alone, and comp_lp+logit)
    float ml = logitA[0];
    float mt = clpA[0] + logitA[0];
    #pragma unroll
    for (int jj = 1; jj < 5; ++jj) {
        ml = fmaxf(ml, logitA[jj]);
        mt = fmaxf(mt, clpA[jj] + logitA[jj]);
    }
    float sl = 0.0f, st = 0.0f;
    #pragma unroll
    for (int jj = 0; jj < 5; ++jj) {
        sl += __expf(logitA[jj] - ml);
        st += __expf(clpA[jj] + logitA[jj] - mt);
    }
    // merge with partner thread (other 5 mixtures), wave32 shuffle
    const float ml2 = __shfl_xor(ml, 1);
    const float sl2 = __shfl_xor(sl, 1);
    const float mt2 = __shfl_xor(mt, 1);
    const float st2 = __shfl_xor(st, 1);
    const float M1 = fmaxf(ml, ml2);
    const float S1 = sl * __expf(ml - M1) + sl2 * __expf(ml2 - M1);
    const float M2 = fmaxf(mt, mt2);
    const float S2 = st * __expf(mt - M2) + st2 * __expf(mt2 - M2);
    const float mix_lp = (M2 + __logf(S2)) - (M1 + __logf(S1));

    if (half == 0) red[pL] = mix_lp;
    __syncthreads();

    // deterministic fixed-order tree reduction over 128 pixels
    #pragma unroll
    for (int s = 64; s > 0; s >>= 1) {
        if (tid < s) red[tid] += red[tid + s];
        __syncthreads();
    }
    if (tid == 0) partials[blockIdx.x] = red[0];
}

// 8 blocks per batch image (1024 px / 128 px-per-block) -> fixed-order sum
__global__ __launch_bounds__(128)
void lm_finalize_kernel(const float* __restrict__ partials,
                        float* __restrict__ out)
{
    const int b = threadIdx.x;   // batch 0..127
    float s = 0.0f;
    #pragma unroll
    for (int i = 0; i < 8; ++i) s += partials[b * 8 + i];
    out[b] = s;
}

extern "C" void kernel_launch(void* const* d_in, const int* in_sizes, int n_in,
                              void* d_out, int out_size, void* d_ws, size_t ws_size,
                              hipStream_t stream) {
    (void)in_sizes; (void)n_in; (void)out_size; (void)ws_size;
    const float* x     = (const float*)d_in[0];   // (128,32,32,256)
    const float* value = (const float*)d_in[1];   // (128,32,32,3)
    const float* Wc    = (const float*)d_in[2];   // (256,100)
    const float* bc    = (const float*)d_in[3];   // (100,)
    float* out = (float*)d_out;                   // (128,)

    // d_ws layout (dwords): [0,14336) packed W, [14336,14448) bias,
    // [14464,15488) block partials  (~62KB total)
    unsigned int* wpack    = (unsigned int*)d_ws;
    float*        bpad     = (float*)d_ws + W_DWORDS;
    float*        partials = (float*)d_ws + 14464;

    lm_pack_kernel<<<56, THREADS, 0, stream>>>(Wc, bc, wpack, bpad);
    lm_main_kernel<<<1024, THREADS, 0, stream>>>(x, value, wpack, bpad, partials);
    lm_finalize_kernel<<<1, 128, 0, stream>>>(partials, out);
}
